// TALHead_55722905698432
// MI455X (gfx1250) — matrix operand order repaired and checked by reading, not compiled, not run on hardware
//
#include <hip/hip_runtime.h>
#include <hip/hip_bf16.h>

typedef __attribute__((ext_vector_type(16))) _Float16 v16h;
typedef __attribute__((ext_vector_type(8)))  float    v8f;

// ---------------------------------------------------------------------------
// 1. fp32 NCHW -> f16 NHWC conversion (coalesced on the NHWC output)
// ---------------------------------------------------------------------------
__global__ void nchw_f32_to_nhwc_f16(const float* __restrict__ in,
                                     _Float16* __restrict__ out,
                                     int B_, int C, int H, int W) {
    const int gid = blockIdx.x * blockDim.x + threadIdx.x;
    const int total = B_ * C * H * W;
    if (gid >= total) return;
    const int c = gid % C;
    int t = gid / C;
    const int x = t % W; t /= W;
    const int y = t % H;
    const int b = t / H;
    out[gid] = (_Float16)in[(size_t)((b * C + c) * H + y) * W + x];
}

// ---------------------------------------------------------------------------
// 2. Pack OIHW fp32 conv weights into WMMA B-fragment order (f16).
//    GEMM K index: k = tap*Cin + c   (tap = ky*T + kx)
//    B layout (16x16x32 f16): lane L holds N = L%16, fragment half j maps to
//    K = (L/16)*16 + j within the 32-wide K-step.
//    dst[((nt*KSTEPS + ks)*32 + lane)*16 + j]
// ---------------------------------------------------------------------------
__global__ void pack_weights_f16(const float* __restrict__ w,
                                 _Float16* __restrict__ dst,
                                 int Cin, int T, int KSTEPS) {
    const int gid = blockIdx.x * blockDim.x + threadIdx.x;
    const int total = KSTEPS << 13;                 // 16 * KSTEPS * 32 * 16
    if (gid >= total) return;
    const int j    = gid & 15;
    const int lane = (gid >> 4) & 31;
    const int ks   = (gid >> 9) % KSTEPS;
    const int nt   = gid / (KSTEPS << 9);
    const int k    = ks * 32 + ((lane >> 4) << 4) + j;
    const int n    = nt * 16 + (lane & 15);
    const int tap  = k / Cin;
    const int c    = k - tap * Cin;
    const int TT   = T * T;
    dst[gid] = (_Float16)w[(size_t)(n * Cin + c) * TT + tap];
}

// ---------------------------------------------------------------------------
// 3. Implicit-GEMM conv (1x1 or 3x3, 'same' padding) + BN(scale,bias) + SiLU.
//    Block = 256 threads = 8 waves; output tile 64(M) x 256(N).
//    Software pipeline: double-buffered LDS A-tiles filled with gfx1250
//    async memory->LDS copies (ASYNCcnt); while WMMA consumes buffer p, the
//    copy for the next K-step is in flight into buffer p^1
//    (s_wait_asynccnt 1 keeps exactly one copy outstanding).
//    Halo rows are zero-filled with a plain ds store under EXEC divergence.
//    lc = log2(Cin); Cin is always a power of two here (256/512/1024).
// ---------------------------------------------------------------------------
template <int T>
__global__ __launch_bounds__(256)
void conv_wmma_silu(const _Float16* __restrict__ actIn,
                    const _Float16* __restrict__ wPack,
                    const float* __restrict__ scale,
                    const float* __restrict__ bias,
                    _Float16* __restrict__ actOut,
                    int H, int W, int lc, int KSTEPS) {
    __shared__ _Float16 lds_a[2][64][48];   // 2 x (64 rows x 32 k-halves, pad 48)

    const int Cin    = 1 << lc;
    const int tid    = threadIdx.x;
    const int lane   = tid & 31;
    const int wv     = tid >> 5;
    const int m_sub  = wv >> 1;
    const int n_half = wv & 1;
    const int m0     = blockIdx.x * 64;

    // staging role: thread -> (row 0..63, 8-half chunk 0..3)
    const int r     = tid >> 2;
    const int cpart = (tid & 3) * 8;
    const int HW    = H * W;
    const int m_r   = m0 + r;
    const int b_r   = m_r / HW;
    const int rem   = m_r - b_r * HW;
    const int y_r   = rem / W;
    const int x_r   = rem - y_r * W;
    const int off   = (T - 1) >> 1;
    // LDS byte offset of this thread's 16B staging chunk (lds_a is at base 0)
    const unsigned ldsbase = (unsigned)r * 96u + (unsigned)cpart * 2u;

    // Issue the A-tile staging for K-step k into LDS buffer p.
    auto stage = [&](int k, int p) {
        const int k0  = k * 32;
        const int tap = k0 >> lc;              // Cin is a power of two
        const int c0  = k0 & (Cin - 1);
        const int ky  = tap / T;               // T is a compile-time constant
        const int kx  = tap - ky * T;
        const int yy  = y_r + ky - off;
        const int xx  = x_r + kx - off;
        if (yy >= 0 && yy < H && xx >= 0 && xx < W) {
            // async memory -> LDS copy (GVS mode: SGPR64 base + VGPR32 offset)
            const unsigned ldso = ldsbase + (unsigned)p * 6144u;
            const unsigned goff =
                (((unsigned)((b_r * H + yy) * W + xx) << lc) +
                 (unsigned)(c0 + cpart)) * 2u;
            asm volatile("global_load_async_to_lds_b128 %0, %1, %2 offset:0"
                         :: "v"(ldso), "v"(goff), "s"(actIn)
                         : "memory");
        } else {
            *(uint4*)&lds_a[p][r][cpart] = make_uint4(0u, 0u, 0u, 0u);
        }
    };

    v8f acc[8] = {};

    // A-fragment addressing: contiguous k = 16*(lane/16) + s, same order as packed B
    const int arow = m_sub * 16 + (lane & 15);
    const int k0l  = (lane & 16) ? 16 : 0;

    stage(0, 0);   // prologue: first tile in flight

    for (int ks = 0; ks < KSTEPS; ++ks) {
        const int p = ks & 1;
        if (ks + 1 < KSTEPS) {
            stage(ks + 1, p ^ 1);                       // next tile in flight
            asm volatile("s_wait_asynccnt 0x1" ::: "memory");  // current done
        } else {
            asm volatile("s_wait_asynccnt 0x0" ::: "memory");
        }
        __syncthreads();   // buffer p fully resident for all waves

        union { v16h v; uint4 q[2]; } af;
        af.q[0] = *(const uint4*)&lds_a[p][arow][k0l];
        af.q[1] = *(const uint4*)&lds_a[p][arow][k0l + 8];

#pragma unroll
        for (int i = 0; i < 8; ++i) {
            const int nt = n_half * 8 + i;
            const _Float16* wp =
                wPack + ((((size_t)nt * KSTEPS + ks) * 32 + lane) << 4);
            union { v16h v; uint4 q[2]; } bf;
            bf.q[0] = ((const uint4*)wp)[0];
            bf.q[1] = ((const uint4*)wp)[1];
            if (ks + 1 < KSTEPS)
                __builtin_prefetch(wp + 512, 0, 0);   // next k-step's fragment
            acc[i] = __builtin_amdgcn_wmma_f32_16x16x32_f16(
                false, af.v, false, bf.v, (short)0, acc[i], false, false);
        }

        __syncthreads();   // everyone done reading buffer p before reuse
    }

    // Epilogue: BN scale/bias + SiLU, store f16 NHWC (C = 256).
    const int mrow = m0 + m_sub * 16 + ((lane >> 4) << 3);
#pragma unroll
    for (int i = 0; i < 8; ++i) {
        const int n   = (n_half * 8 + i) * 16 + (lane & 15);
        const float s  = scale[n];
        const float bb = bias[n];
#pragma unroll
        for (int e = 0; e < 8; ++e) {
            float y = acc[i][e] * s + bb;
            y = y / (1.0f + __expf(-y));              // SiLU
            actOut[(size_t)(mrow + e) * 256 + n] = (_Float16)y;
        }
    }
}

// ---------------------------------------------------------------------------
// 4. Fused 1x1 preds (cls 80 / reg 4 / obj 1) + sigmoid + grid/stride decode.
//    Writes directly into out[B, 8400, 85] (fp32).
// ---------------------------------------------------------------------------
__global__ void pred_decode(const _Float16* __restrict__ clsF,
                            const _Float16* __restrict__ regF,
                            const float* __restrict__ cw, const float* __restrict__ cb,
                            const float* __restrict__ rw, const float* __restrict__ rb,
                            const float* __restrict__ ow, const float* __restrict__ ob,
                            float* __restrict__ out,
                            int B_, int H, int W, int strideLvl, int anchorBase) {
    const int gid = blockIdx.x * blockDim.x + threadIdx.x;
    const int total = B_ * H * W * 85;
    if (gid >= total) return;
    const int ch  = gid % 85;
    const int pos = gid / 85;
    const int x = pos % W;
    int t = pos / W;
    const int y = t % H;
    const int b = t / H;

    const _Float16* feat;
    const float* wrow;
    float acc;
    if (ch < 4)       { feat = regF; wrow = rw + ch * 256;       acc = rb[ch];     }
    else if (ch == 4) { feat = regF; wrow = ow;                  acc = ob[0];      }
    else              { feat = clsF; wrow = cw + (ch - 5) * 256; acc = cb[ch - 5]; }

    const _Float16* f = feat + (size_t)pos * 256;
    for (int c = 0; c < 256; ++c) acc += (float)f[c] * wrow[c];

    float v;
    if (ch == 0)      v = (acc + (float)x) * (float)strideLvl;
    else if (ch == 1) v = (acc + (float)y) * (float)strideLvl;
    else if (ch < 4)  v = __expf(acc) * (float)strideLvl;
    else              v = 1.0f / (1.0f + __expf(-acc));

    const int anchor = anchorBase + y * W + x;
    out[((size_t)b * 8400 + anchor) * 85 + ch] = v;
}

// ---------------------------------------------------------------------------
// Host-side orchestration
// ---------------------------------------------------------------------------
extern "C" void kernel_launch(void* const* d_in, const int* in_sizes, int n_in,
                              void* d_out, int out_size, void* d_ws, size_t ws_size,
                              hipStream_t stream) {
    (void)in_sizes; (void)n_in; (void)out_size; (void)ws_size;

    const int Bn = 8;
    static const int CIN[3]   = {256, 512, 1024};
    static const int LC[3]    = {8, 9, 10};
    static const int HH[3]    = {80, 40, 20};
    static const int STR[3]   = {8, 16, 32};
    static const int ABASE[3] = {0, 6400, 8000};

    const size_t ACT_HALVES = (size_t)8 * 80 * 80 * 256;   // max per-level NHWC f16
    _Float16* inF16 = (_Float16*)d_ws;
    _Float16* bufA  = inF16 + ACT_HALVES;   // stem out
    _Float16* bufB  = bufA  + ACT_HALVES;   // branch temp
    _Float16* bufC  = bufB  + ACT_HALVES;   // cls feature
    _Float16* bufD  = bufC  + ACT_HALVES;   // reg feature
    _Float16* wStem = bufD  + ACT_HALVES;   // up to 1024*256 halves
    _Float16* wC1   = wStem + 1024 * 256;
    _Float16* wC2   = wC1   + 2304 * 256;
    _Float16* wC3   = wC2   + 2304 * 256;
    _Float16* wC4   = wC3   + 2304 * 256;

    float* out = (float*)d_out;
    const dim3 blk(256);

    for (int L = 0; L < 3; ++L) {
        const int Cin = CIN[L], H = HH[L], W = HH[L];
        const int M = Bn * H * W;
        const int base = 3 + L * 21;

        const float* xin    = (const float*)d_in[L];
        const float* stem_w = (const float*)d_in[base + 0];
        const float* stem_s = (const float*)d_in[base + 1];
        const float* stem_b = (const float*)d_in[base + 2];
        const float* c1_w = (const float*)d_in[base + 3];
        const float* c1_s = (const float*)d_in[base + 4];
        const float* c1_b = (const float*)d_in[base + 5];
        const float* c2_w = (const float*)d_in[base + 6];
        const float* c2_s = (const float*)d_in[base + 7];
        const float* c2_b = (const float*)d_in[base + 8];
        const float* r1_w = (const float*)d_in[base + 9];
        const float* r1_s = (const float*)d_in[base + 10];
        const float* r1_b = (const float*)d_in[base + 11];
        const float* r2_w = (const float*)d_in[base + 12];
        const float* r2_s = (const float*)d_in[base + 13];
        const float* r2_b = (const float*)d_in[base + 14];
        const float* cls_w = (const float*)d_in[base + 15];
        const float* cls_b = (const float*)d_in[base + 16];
        const float* reg_w = (const float*)d_in[base + 17];
        const float* reg_b = (const float*)d_in[base + 18];
        const float* obj_w = (const float*)d_in[base + 19];
        const float* obj_b = (const float*)d_in[base + 20];

        // layout + precision conversion of the input feature map
        {
            const int total = M * Cin;
            nchw_f32_to_nhwc_f16<<<dim3((total + 255) / 256), blk, 0, stream>>>(
                xin, inF16, Bn, Cin, H, W);
        }

        // weight packing (fragment-major f16)
        const int KS_STEM = Cin / 32;
        const int KS_33   = 2304 / 32;   // 72
        pack_weights_f16<<<dim3((8192 * KS_STEM + 255) / 256), blk, 0, stream>>>(stem_w, wStem, Cin, 1, KS_STEM);
        pack_weights_f16<<<dim3((8192 * KS_33 + 255) / 256), blk, 0, stream>>>(c1_w, wC1, 256, 3, KS_33);
        pack_weights_f16<<<dim3((8192 * KS_33 + 255) / 256), blk, 0, stream>>>(c2_w, wC2, 256, 3, KS_33);
        pack_weights_f16<<<dim3((8192 * KS_33 + 255) / 256), blk, 0, stream>>>(r1_w, wC3, 256, 3, KS_33);
        pack_weights_f16<<<dim3((8192 * KS_33 + 255) / 256), blk, 0, stream>>>(r2_w, wC4, 256, 3, KS_33);

        // WMMA conv pipeline (M is a multiple of 64 for all levels)
        const dim3 g(M / 64);
        conv_wmma_silu<1><<<g, blk, 0, stream>>>(inF16, wStem, stem_s, stem_b, bufA, H, W, LC[L], KS_STEM);
        conv_wmma_silu<3><<<g, blk, 0, stream>>>(bufA,  wC1,   c1_s,   c1_b,   bufB, H, W, 8, KS_33);
        conv_wmma_silu<3><<<g, blk, 0, stream>>>(bufB,  wC2,   c2_s,   c2_b,   bufC, H, W, 8, KS_33);
        conv_wmma_silu<3><<<g, blk, 0, stream>>>(bufA,  wC3,   r1_s,   r1_b,   bufB, H, W, 8, KS_33);
        conv_wmma_silu<3><<<g, blk, 0, stream>>>(bufB,  wC4,   r2_s,   r2_b,   bufD, H, W, 8, KS_33);

        // fused preds + sigmoid + decode
        {
            const int total = M * 85;
            pred_decode<<<dim3((total + 255) / 256), blk, 0, stream>>>(
                bufC, bufD, cls_w, cls_b, reg_w, reg_b, obj_w, obj_b,
                out, Bn, H, W, STR[L], ABASE[L]);
        }
    }
}
